// OrdinalRegression_23055384445147
// MI455X (gfx1250) — compile-verified, matrix operand order checked by reading, not run on hardware
//
#include <hip/hip_runtime.h>
#include <hip/hip_bf16.h>

// ---------------------------------------------------------------------------
// OrdinalRegression on MI455X (gfx1250, wave32).
//
// Cost model: 1024x1024x128 pairwise L1 (~270M VALU ops) dominates; s-matrix
// (4MB) + stats live entirely in the 192MB L2, so HBM bandwidth is a non-issue.
// The L1 distance cannot be expressed as a matmul, so the main pass is VALU;
// exp() runs on the TRANS pipe; the global reductions (sum(u), sum(u*s)) are
// funneled through v_wmma_f32_16x16x32_f16 (16-bit XDL WMMA co-executes with
// VALU, CDNA5 ISA 8.4). Tile staging into LDS uses the CDNA5 async path
// (global_load_async_to_lds_b128 + s_wait_asynccnt): no VGPR round-trip,
// single buffer: each tile's copy is waited on before its |diff| VALU work.
// ---------------------------------------------------------------------------

typedef __attribute__((ext_vector_type(16))) _Float16 v16h;
typedef __attribute__((ext_vector_type(8)))  float    v8f;

#define NXD 1024
#define NYD 1024
#define EDIM 128
#define LDSS 132   // padded LDS row stride (floats): 132 % 64 = 4 -> conflict-free
                   // row stride in bytes = 528 (16B aligned -> B128-legal)

// Async global -> LDS, 16 bytes, tracked by ASYNCcnt.
__device__ __forceinline__ void async_g2l_b128(unsigned lds_byte_off,
                                               const void* gaddr) {
  asm volatile("global_load_async_to_lds_b128 %0, %1, off"
               :
               : "v"(lds_byte_off), "v"((unsigned long long)(size_t)gaddr)
               : "memory");
}

__device__ __forceinline__ void wait_async0() {
  asm volatile("s_wait_asynccnt 0" ::: "memory");
}

// Flat address of a __shared__ object keeps the LDS byte offset in addr[31:0].
__device__ __forceinline__ unsigned lds_off(const void* p) {
  return (unsigned)(size_t)p;
}

// ---------------------------------------------------------------------------
// Kernel 1: 64 workgroups, each owns 16 rows of s.
// s[i,j] = -sum_d |zx[i,d]-zy[j,d]|, rowmax[i], rowsum[i] = sum_j exp(s-rowmax).
// Also zeroes the two global accumulators (block 0).
// ---------------------------------------------------------------------------
__global__ __launch_bounds__(256) void k_rows(const float* __restrict__ zx,
                                              const float* __restrict__ zy,
                                              float* __restrict__ s,
                                              float* __restrict__ rowmax,
                                              float* __restrict__ rowsum,
                                              float* __restrict__ gacc) {
  __shared__ __align__(16) float lx[16 * LDSS];
  __shared__ __align__(16) float ly[16 * LDSS];
  __shared__ float red[256];
  __shared__ float rm_sh[16];

  const int t     = threadIdx.x;
  const int i_loc = t >> 4;    // 0..15 : local row
  const int jl    = t & 15;    // 0..15 : local col
  const int i0    = blockIdx.x * 16;

  if (blockIdx.x == 0 && t < 2) gacc[t] = 0.0f;   // zero sum_u / sum_us

  const unsigned lx_base = lds_off(lx);
  const unsigned ly_base = lds_off(ly);

  // zx tile -> LDS: 512 x 16B async chunks (chunk c -> row c>>5, 16B col (c&31))
  for (int c = t; c < 512; c += 256)
    async_g2l_b128(lx_base + (unsigned)((c >> 5) * (LDSS * 4) + (c & 31) * 16),
                   zx + i0 * EDIM + c * 4);

  float m = -3.402823466e38f;
  for (int jb = 0; jb < NYD; jb += 16) {
    __syncthreads();   // previous tile fully consumed before overwrite
    for (int c = t; c < 512; c += 256) {
      async_g2l_b128(ly_base + (unsigned)((c >> 5) * (LDSS * 4) + (c & 31) * 16),
                     zy + jb * EDIM + c * 4);
      if (jb + 16 < NYD)
        __builtin_prefetch(zy + (jb + 16) * EDIM + c * 4, 0, 3);  // global_prefetch_b8
    }
    wait_async0();     // LDS writes of this tile complete (covers lx on iter 0)
    __syncthreads();

    const float* xr = &lx[i_loc * LDSS];
    const float* yr = &ly[jl * LDSS];
    float acc = 0.0f;
#pragma unroll
    for (int d = 0; d < EDIM; ++d)
      acc += fabsf(xr[d] - yr[d]);     // v_sub + v_add with |src| modifier

    const float sv = -acc;
    s[(i0 + i_loc) * NYD + jb + jl] = sv;
    m = fmaxf(m, sv);
  }

  // rowmax: reduce across the 16 threads of each row
  red[t] = m;
  __syncthreads();
  if (jl == 0) {
    float mm = red[i_loc * 16];
    for (int k = 1; k < 16; ++k) mm = fmaxf(mm, red[i_loc * 16 + k]);
    rm_sh[i_loc] = mm;
    rowmax[i0 + i_loc] = mm;
  }
  __syncthreads();

  // rowsum of exp(s - rowmax); s re-read from L2-resident workspace
  const float rm = rm_sh[i_loc];
  float se = 0.0f;
  for (int jb = 0; jb < NYD; jb += 16)
    se += __expf(s[(i0 + i_loc) * NYD + jb + jl] - rm);
  red[t] = se;
  __syncthreads();
  if (jl == 0) {
    float ss = 0.0f;
    for (int k = 0; k < 16; ++k) ss += red[i_loc * 16 + k];
    rowsum[i0 + i_loc] = ss;
  }
}

// ---------------------------------------------------------------------------
// Kernel 2: 64 workgroups, each owns 16 columns. colmax / colsum of exp.
// ---------------------------------------------------------------------------
__global__ __launch_bounds__(256) void k_cols(const float* __restrict__ s,
                                              float* __restrict__ colmax,
                                              float* __restrict__ colsum) {
  __shared__ float red[256];
  __shared__ float cm_sh[16];

  const int t  = threadIdx.x;
  const int ir = t >> 4;
  const int jl = t & 15;
  const int j  = blockIdx.x * 16 + jl;

  float m = -3.402823466e38f;
  for (int ib = 0; ib < NXD; ib += 16)
    m = fmaxf(m, s[(ib + ir) * NYD + j]);
  red[t] = m;
  __syncthreads();
  if (ir == 0) {
    float mm = red[jl];
    for (int k = 1; k < 16; ++k) mm = fmaxf(mm, red[k * 16 + jl]);
    cm_sh[jl] = mm;
    colmax[j] = mm;
  }
  __syncthreads();

  const float cmv = cm_sh[jl];
  float se = 0.0f;
  for (int ib = 0; ib < NXD; ib += 16)
    se += __expf(s[(ib + ir) * NYD + j] - cmv);
  red[t] = se;
  __syncthreads();
  if (ir == 0) {
    float ss = 0.0f;
    for (int k = 0; k < 16; ++k) ss += red[k * 16 + jl];
    colsum[j] = ss;
  }
}

// ---------------------------------------------------------------------------
// Kernel 3: 32 blocks x 8 waves = 256 waves; 16 tiles (16x16) per wave.
// u = a + b - a*b ; accumulate sum(u) and sum(u*s) through the matrix pipe:
// one v_wmma_f32_16x16x32_f16 per accumulator per tile (A = values, B = ones).
// Each value occupies exactly one A slot, so D holds exact f32 partial sums.
// ---------------------------------------------------------------------------
__global__ __launch_bounds__(256) void k_accum(const float* __restrict__ s,
                                               const float* __restrict__ rowmax,
                                               const float* __restrict__ rowsum,
                                               const float* __restrict__ colmax,
                                               const float* __restrict__ colsum,
                                               float* __restrict__ gacc) {
  __shared__ float rm[NXD], rs[NXD], cm[NYD], cs[NYD];
  const int t = threadIdx.x;
  for (int k = t; k < NXD; k += 256) {
    rm[k] = rowmax[k];
    rs[k] = rowsum[k];
    cm[k] = colmax[k];
    cs[k] = colsum[k];
  }
  __syncthreads();

  const int lane    = t & 31;
  const int wave    = t >> 5;
  const int wglobal = blockIdx.x * 8 + wave;
  const int half    = lane >> 4;
  const int n       = lane & 15;

  v16h ones;
#pragma unroll
  for (int v = 0; v < 16; ++v) ones[v] = (_Float16)1.0f;

  v8f accU  = {};
  v8f accUS = {};

  for (int k = 0; k < 16; ++k) {           // uniform trip count: EXEC stays all-1s
    const int tile = wglobal * 16 + k;     // 0..4095
    const int i0   = (tile >> 6) * 16;
    const int j    = (tile & 63) * 16 + n;
    const float cmv = cm[j];
    const float csv = cs[j];

    v16h Au, Aus;
#pragma unroll
    for (int v = 0; v < 8; ++v) {
      const int   i  = i0 + v + 8 * half;
      const float sv = s[i * NYD + j];
      const float a  = __expf(sv - rm[i]) / rs[i];
      const float b  = __expf(sv - cmv) / csv;
      const float u  = a + b - a * b;
      Au[v]      = (_Float16)u;
      Aus[v]     = (_Float16)(u * sv);
      Au[v + 8]  = (_Float16)0.0f;
      Aus[v + 8] = (_Float16)0.0f;
    }

    accU  = __builtin_amdgcn_wmma_f32_16x16x32_f16(
                false, Au,  false, ones, (short)0, accU,  false, false);
    accUS = __builtin_amdgcn_wmma_f32_16x16x32_f16(
                false, Aus, false, ones, (short)0, accUS, false, false);
  }

  // D[m,n] = per-row partial sums (same for every n). Lanes 0-15 carry rows
  // 0-7 in their 8 VGPRs, lanes 16-31 rows 8-15 -> sum 8 VGPRs, add l^16.
  float su = 0.0f, sus = 0.0f;
#pragma unroll
  for (int v = 0; v < 8; ++v) { su += accU[v]; sus += accUS[v]; }
  su  += __shfl_xor(su, 16, 32);
  sus += __shfl_xor(sus, 16, 32);

  if (lane == 0) {
    atomicAdd(&gacc[0], su);    // global_atomic_add_f32
    atomicAdd(&gacc[1], sus);
  }
}

// ---------------------------------------------------------------------------
// Kernel 4: c = sum(u*s)/sum(u); out[k] = c*theta[k] + beta[k]
// ---------------------------------------------------------------------------
__global__ void k_final(const float* __restrict__ gacc,
                        const float* __restrict__ theta,
                        const float* __restrict__ beta,
                        float* __restrict__ out) {
  const int t = threadIdx.x;
  if (t < 4) {
    const float c = gacc[1] / gacc[0];
    out[t] = c * theta[t] + beta[t];
  }
}

// ---------------------------------------------------------------------------
extern "C" void kernel_launch(void* const* d_in, const int* in_sizes, int n_in,
                              void* d_out, int out_size, void* d_ws, size_t ws_size,
                              hipStream_t stream) {
  (void)in_sizes; (void)n_in; (void)out_size; (void)ws_size;
  const float* zx    = (const float*)d_in[0];  // (1024, 128) f32
  const float* zy    = (const float*)d_in[1];  // (1024, 128) f32
  const float* theta = (const float*)d_in[2];  // (1, 4) f32
  const float* beta  = (const float*)d_in[3];  // (4,) f32

  float* ws     = (float*)d_ws;
  float* s      = ws;                       // 1024*1024
  float* rowmax = ws + NXD * NYD;           // 1024
  float* rowsum = rowmax + NXD;             // 1024
  float* colmax = rowsum + NXD;             // 1024
  float* colsum = colmax + NYD;             // 1024
  float* gacc   = colsum + NYD;             // 2: sum_u, sum_us

  k_rows <<<64, 256, 0, stream>>>(zx, zy, s, rowmax, rowsum, gacc);
  k_cols <<<64, 256, 0, stream>>>(s, colmax, colsum);
  k_accum<<<32, 256, 0, stream>>>(s, rowmax, rowsum, colmax, colsum, gacc);
  k_final<<<1, 32, 0, stream>>>(gacc, theta, beta, (float*)d_out);
}
